// Encoder_48481590837659
// MI455X (gfx1250) — compile-verified, matrix operand order checked by reading, not run on hardware
//
#include <hip/hip_runtime.h>
#include <hip/hip_bf16.h>

typedef __attribute__((ext_vector_type(16))) _Float16 v16h;
typedef __attribute__((ext_vector_type(8)))  _Float16 v8h;
typedef __attribute__((ext_vector_type(8)))  float    v8f;

#define BATCH 16
#define NPTS  2048
#define BN    (BATCH * NPTS)
#define KNN_K 16

// ---------------------------------------------------------------------------
// CDNA5 16-bit A/B fragment (ISA 7.12.2): lane (idx = lane&15, half = lane>>4)
// holds two contiguous 8-element f16 runs of row `idx`:
//   f[0..7]  = row[koff + half*8      .. +8)
//   f[8..15] = row[koff + 16 + half*8 .. +8)
// => two 128-bit loads per fragment, no shuffling.
// ---------------------------------------------------------------------------
__device__ __forceinline__ v16h load_frag_f16(const _Float16* __restrict__ base,
                                              int ld, int idx, int koff, int half) {
    const _Float16* p = base + (size_t)idx * ld + koff + half * 8;
    v8h c0 = *(const v8h*)p;
    v8h c1 = *(const v8h*)(p + 16);
    v16h f;
#pragma unroll
    for (int j = 0; j < 8; ++j) { f[j] = c0[j]; f[8 + j] = c1[j]; }
    return f;
}

__device__ __forceinline__ v16h load_frag_f32(const float* __restrict__ base,
                                              int ld, int idx, int koff, int half) {
    const float* p = base + (size_t)idx * ld + koff + half * 8;
    v16h f;
#pragma unroll
    for (int j = 0; j < 8; ++j) { f[j] = (_Float16)p[j]; f[8 + j] = (_Float16)p[16 + j]; }
    return f;
}

__device__ __forceinline__ float atomicMaxFloat(float* addr, float val) {
    int* ia = (int*)addr;
    int old = __float_as_int(*addr);
    while (__int_as_float(old) < val) {
        int assumed = old;
        old = atomicCAS(ia, assumed, __float_as_int(val));
        if (old == assumed) break;
    }
    return __int_as_float(old);
}

// ---------------------------------------------------------------------------
// WMMA GEMM: out[M, Npad] = act((A[M,ldA](f32) x Wt^T(f16)) * scale + shift)
// One wave per block computes a 16 x (16*NT) strip; NT independent
// accumulators keep the XDL pipe busy. All loads unconditional (padded
// buffers, zero-padded weights). Optional fused row max-pool epilogue.
// ---------------------------------------------------------------------------
template <int NT>
__global__ __launch_bounds__(32)
void wmma_gemm(const float* __restrict__ A, int ldA,
               const _Float16* __restrict__ Wt, int kpad,
               const float* __restrict__ scale, const float* __restrict__ shift,
               float* __restrict__ out, int ldOut, float* __restrict__ poolOut,
               int kChunks, int relu, int pool_rows, int Nout) {
    const int lane = threadIdx.x;
    const int idx  = lane & 15;
    const int half = lane >> 4;
    const int mBase = blockIdx.x * 16;
    const int nBase = blockIdx.y * 16 * NT;

    v8f acc[NT];
#pragma unroll
    for (int t = 0; t < NT; ++t) acc[t] = (v8f){};

    const float* Abase = A + (size_t)mBase * ldA;
    for (int kc = 0; kc < kChunks; ++kc) {
        const int koff = kc * 32;
        if (kc + 1 < kChunks)
            __builtin_prefetch(Abase + (size_t)idx * ldA + koff + 32, 0, 1);
        v16h af = load_frag_f32(Abase, ldA, idx, koff, half);
#pragma unroll
        for (int t = 0; t < NT; ++t) {
            v16h bf = load_frag_f16(Wt + (size_t)(nBase + t * 16) * kpad,
                                    kpad, idx, koff, half);
            acc[t] = __builtin_amdgcn_wmma_f32_16x16x32_f16(false, af, false, bf,
                                                            (short)0, acc[t], false, false);
        }
    }

#pragma unroll
    for (int t = 0; t < NT; ++t) {
        const int gn = nBase + t * 16 + idx;   // C/D: column = lane&15
        const float sc = scale[gn];
        const float sh = shift[gn];
#pragma unroll
        for (int v = 0; v < 8; ++v) {          // rows m = v + 8*half
            const int gm = mBase + v + 8 * half;
            float tv = acc[t][v] * sc + sh;
            if (relu) tv = fmaxf(tv, 0.f);
            if (pool_rows)
                atomicMaxFloat(&poolOut[(size_t)(gm / pool_rows) * Nout + gn], tv);
            else
                out[(size_t)gm * ldOut + gn] = tv;
        }
    }
}

// ---------------------------------------------------------------------------
// Fused KNN on padded f16 features hH[BN][CPAD]. One wave owns 16 query rows.
// Query fragments live in registers for the whole sweep. Column tiles are
// processed in PAIRS (two independent WMMA accumulator chains interleaved to
// hide XDL latency). Distances land in a 16x512 LDS strip; ALL 32 lanes scan
// (lane = (row, column-half), float4 LDS loads) keeping register-resident
// sorted top-16 partials; a final lexicographic (d,idx) 2-list merge exactly
// reproduces top_k's stable ascending order.
// ---------------------------------------------------------------------------
template <int CPAD>
__global__ __launch_bounds__(32)
void knn_kernel(const _Float16* __restrict__ hH, const float* __restrict__ sq,
                int* __restrict__ knnIdx) {
    constexpr int CHUNK = 512;
    constexpr int KC = CPAD / 32;
    __shared__ float dS[16 * CHUNK];
    __shared__ float mD[32 * KNN_K];
    __shared__ int   mI[32 * KNN_K];

    const int lane = threadIdx.x;
    const int idx  = lane & 15;
    const int half = lane >> 4;
    const int b       = blockIdx.x / (NPTS / 16);
    const int rowBase = (blockIdx.x % (NPTS / 16)) * 16;
    const _Float16* hb  = hH + (size_t)b * NPTS * CPAD;
    const float*    sqb = sq + (size_t)b * NPTS;

    v16h afrag[KC];
#pragma unroll
    for (int kc = 0; kc < KC; ++kc)
        afrag[kc] = load_frag_f16(hb + (size_t)rowBase * CPAD, CPAD, idx, kc * 32, half);

    float sqm[8];
#pragma unroll
    for (int v = 0; v < 8; ++v) sqm[v] = sqb[rowBase + v + 8 * half];

    float bd[KNN_K];
    int   bi[KNN_K];
#pragma unroll
    for (int i = 0; i < KNN_K; ++i) { bd[i] = 3.4e38f; bi[i] = 0; }

    for (int chunk = 0; chunk < NPTS / CHUNK; ++chunk) {
        const int chunkBase = chunk * CHUNK;
        for (int t = 0; t < CHUNK / 16; t += 2) {
            const int col0 = chunkBase + t * 16;
            const int col1 = col0 + 16;
            const _Float16* b0 = hb + (size_t)col0 * CPAD;
            const _Float16* b1 = hb + (size_t)col1 * CPAD;
            __builtin_prefetch(b1 + 16 * CPAD + (size_t)idx * CPAD, 0, 1);
            v8f a0 = (v8f){};
            v8f a1 = (v8f){};
#pragma unroll
            for (int kc = 0; kc < KC; ++kc) {   // two independent WMMA chains
                v16h bf0 = load_frag_f16(b0, CPAD, idx, kc * 32, half);
                v16h bf1 = load_frag_f16(b1, CPAD, idx, kc * 32, half);
                a0 = __builtin_amdgcn_wmma_f32_16x16x32_f16(false, afrag[kc], false, bf0,
                                                            (short)0, a0, false, false);
                a1 = __builtin_amdgcn_wmma_f32_16x16x32_f16(false, afrag[kc], false, bf1,
                                                            (short)0, a1, false, false);
            }
            const float sq0 = sqb[col0 + idx];
            const float sq1 = sqb[col1 + idx];
#pragma unroll
            for (int v = 0; v < 8; ++v) {
                const int m = v + 8 * half;
                dS[m * CHUNK + t * 16 + idx]        = sqm[v] + sq0 - 2.f * a0[v];
                dS[m * CHUNK + (t + 1) * 16 + idx]  = sqm[v] + sq1 - 2.f * a1[v];
            }
        }
        __syncthreads();
        {   // full-wave scan: lane handles row=idx, columns [half*256, +256)
            const int j0 = half * (CHUNK / 2);
            const float* rowS = &dS[idx * CHUNK];
            for (int j = j0; j < j0 + CHUNK / 2; j += 4) {
                float4 c4 = *(const float4*)&rowS[j];
                float cv[4] = {c4.x, c4.y, c4.z, c4.w};
#pragma unroll
                for (int e = 0; e < 4; ++e) {
                    float cd = cv[e];
                    if (cd < bd[KNN_K - 1]) {
                        float vk = cd; int ik = chunkBase + j + e;
#pragma unroll
                        for (int q = 0; q < KNN_K; ++q) {
                            if (vk < bd[q]) {
                                float tf = bd[q]; bd[q] = vk; vk = tf;
                                int   ti = bi[q]; bi[q] = ik; ik = ti;
                            }
                        }
                    }
                }
            }
        }
        __syncthreads();
    }

    // merge the two per-row partial lists (lexicographic on (d, idx))
#pragma unroll
    for (int q = 0; q < KNN_K; ++q) {
        mD[lane * KNN_K + q] = bd[q];
        mI[lane * KNN_K + q] = bi[q];
    }
    __syncthreads();
    if (lane < 16) {
        const float* d0 = &mD[lane * KNN_K];        const int* i0 = &mI[lane * KNN_K];
        const float* d1 = &mD[(lane + 16) * KNN_K]; const int* i1 = &mI[(lane + 16) * KNN_K];
        int p0 = 0, p1 = 0;
        int* outp = &knnIdx[((size_t)b * NPTS + rowBase + lane) * KNN_K];
#pragma unroll
        for (int q = 0; q < KNN_K; ++q) {
            float a = d0[p0], c = d1[p1];
            bool take0 = (a < c) || (a == c && i0[p0] < i1[p1]);
            outp[q] = take0 ? i0[p0] : i1[p1];
            if (take0) ++p0; else ++p1;
        }
    }
}

// ---------------------------------------------------------------------------
// Prep + helper kernels (bandwidth-trivial)
// ---------------------------------------------------------------------------
__global__ void prep_wt(const float* __restrict__ W, _Float16* __restrict__ Wt,
                        int K, int N, int kpad, int npad) {
    int g = blockIdx.x * blockDim.x + threadIdx.x;
    if (g >= npad * kpad) return;
    int n = g / kpad, k = g % kpad;
    Wt[g] = (n < N && k < K) ? (_Float16)W[(size_t)k * N + n] : (_Float16)0.f;
}

__global__ void prep_ss(const float* __restrict__ bias,
                        const float* __restrict__ g, const float* __restrict__ bb,
                        const float* __restrict__ m, const float* __restrict__ v,
                        float* __restrict__ scale, float* __restrict__ shift,
                        int n, int npad) {
    int i = blockIdx.x * blockDim.x + threadIdx.x;
    if (i >= npad) return;
    if (i < n) {
        float s = 1.f, sh = 0.f;
        if (g) { s = g[i] * rsqrtf(v[i] + 1e-3f); sh = bb[i] - m[i] * s; }
        float bv = bias ? bias[i] : 0.f;
        scale[i] = s;
        shift[i] = bv * s + sh;
    } else {
        scale[i] = 0.f;
        shift[i] = 0.f;
    }
}

__global__ void cvtpad_kernel(const float* __restrict__ h, _Float16* __restrict__ out,
                              int C, int CPAD) {
    int g = blockIdx.x * blockDim.x + threadIdx.x;
    if (g >= BN * CPAD) return;
    int i = g / CPAD, c = g % CPAD;
    out[g] = (c < C) ? (_Float16)h[(size_t)i * C + c] : (_Float16)0.f;
}

__global__ void sq_kernel(const float* __restrict__ h, float* __restrict__ sq,
                          int n, int C) {
    int i = blockIdx.x * blockDim.x + threadIdx.x;
    if (i >= n) return;
    float s = 0.f;
    for (int c = 0; c < C; ++c) { float v = h[(size_t)i * C + c]; s += v * v; }
    sq[i] = s;
}

__global__ void cov_kernel(const float* __restrict__ x, const int* __restrict__ idx,
                           float* __restrict__ h0) {   // h0: [BN][32], cols 12..31 zeroed
    int i = blockIdx.x * blockDim.x + threadIdx.x;
    if (i >= BN) return;
    int b = i >> 11;
    const float* xb = x + (size_t)b * NPTS * 3;
    const int*   id = idx + (size_t)i * KNN_K;
    float S0 = 0, S1 = 0, S2 = 0;
    float P00 = 0, P01 = 0, P02 = 0, P11 = 0, P12 = 0, P22 = 0;
#pragma unroll
    for (int k = 0; k < KNN_K; ++k) {
        int j = id[k];
        float p0 = xb[j * 3], p1 = xb[j * 3 + 1], p2 = xb[j * 3 + 2];
        S0 += p0; S1 += p1; S2 += p2;
        P00 += p0 * p0; P01 += p0 * p1; P02 += p0 * p2;
        P11 += p1 * p1; P12 += p1 * p2; P22 += p2 * p2;
    }
    const float invK = 1.f / (float)KNN_K;
    float m0 = S0 * invK, m1 = S1 * invK, m2 = S2 * invK;
    float c00 = P00 - (float)KNN_K * m0 * m0;
    float c01 = P01 - (float)KNN_K * m0 * m1;
    float c02 = P02 - (float)KNN_K * m0 * m2;
    float c11 = P11 - (float)KNN_K * m1 * m1;
    float c12 = P12 - (float)KNN_K * m1 * m2;
    float c22 = P22 - (float)KNN_K * m2 * m2;
    float* o = h0 + (size_t)i * 32;
    o[0] = x[(size_t)i * 3]; o[1] = x[(size_t)i * 3 + 1]; o[2] = x[(size_t)i * 3 + 2];
    o[3] = c00; o[4]  = c01; o[5]  = c02;
    o[6] = c01; o[7]  = c11; o[8]  = c12;
    o[9] = c02; o[10] = c12; o[11] = c22;
#pragma unroll
    for (int c = 12; c < 32; ++c) o[c] = 0.f;
}

__global__ void gmax_kernel(const float* __restrict__ h, const int* __restrict__ idx,
                            float* __restrict__ out, int C) {
    int g = blockIdx.x * blockDim.x + threadIdx.x;
    if (g >= BN * C) return;
    int i = g / C, c = g % C;
    int b = i >> 11;
    const float* hb = h + (size_t)b * NPTS * C;
    const int*   id = idx + (size_t)i * KNN_K;
    float m = -3.4e38f;
#pragma unroll
    for (int k = 0; k < KNN_K; ++k) m = fmaxf(m, hb[(size_t)id[k] * C + c]);
    out[(size_t)i * C + c] = m;
}

__global__ void fill_kernel(float* __restrict__ p, float v, int n) {
    int i = blockIdx.x * blockDim.x + threadIdx.x;
    if (i < n) p[i] = v;
}

// ---------------------------------------------------------------------------
extern "C" void kernel_launch(void* const* d_in, const int* in_sizes, int n_in,
                              void* d_out, int out_size, void* d_ws, size_t ws_size,
                              hipStream_t stream) {
    (void)in_sizes; (void)n_in; (void)out_size; (void)ws_size;
    const float* x       = (const float*)d_in[0];
    const float* conv1_w = (const float*)d_in[1];  const float* conv1_b = (const float*)d_in[2];
    const float* conv2_w = (const float*)d_in[3];  const float* conv2_b = (const float*)d_in[4];
    const float* conv3_w = (const float*)d_in[5];  const float* conv3_b = (const float*)d_in[6];
    const float* g1lw    = (const float*)d_in[7];  const float* g1lb    = (const float*)d_in[8];
    const float* g1cw    = (const float*)d_in[9];  const float* g1cb    = (const float*)d_in[10];
    const float* g2lw    = (const float*)d_in[11]; const float* g2lb    = (const float*)d_in[12];
    const float* g2cw    = (const float*)d_in[13]; const float* g2cb    = (const float*)d_in[14];
    const float* conv4_w = (const float*)d_in[15]; const float* conv4_b = (const float*)d_in[16];
    const float* conv5_w = (const float*)d_in[17]; const float* conv5_b = (const float*)d_in[18];
    const float* bn1g = (const float*)d_in[19]; const float* bn1b = (const float*)d_in[20];
    const float* bn1m = (const float*)d_in[21]; const float* bn1v = (const float*)d_in[22];
    const float* bn2g = (const float*)d_in[23]; const float* bn2b = (const float*)d_in[24];
    const float* bn2m = (const float*)d_in[25]; const float* bn2v = (const float*)d_in[26];
    const float* bn3g = (const float*)d_in[27]; const float* bn3b = (const float*)d_in[28];
    const float* bn3m = (const float*)d_in[29]; const float* bn3v = (const float*)d_in[30];

    // ---- workspace carve-up ----
    char* ws = (char*)d_ws;
    auto carve = [&](size_t bytes) { char* p = ws; ws += (bytes + 255) & ~(size_t)255; return p; };
    float*     sqv  = (float*)carve((size_t)BN * 4);
    int*       idx  = (int*)carve((size_t)BN * KNN_K * 4);
    float*     t0   = (float*)carve((size_t)BN * 128 * 4);
    float*     t1   = (float*)carve((size_t)BN * 128 * 4);
    float*     t2   = (float*)carve((size_t)BN * 128 * 4);
    float*     pool = (float*)carve((size_t)BATCH * 1024 * 4);
    _Float16*  hH   = (_Float16*)carve((size_t)BN * 128 * 2);
    struct L { int K, N, kpad, npad; const float *W, *bias, *g, *b, *m, *v;
               _Float16* wt; float *sc, *sh; };
    L ly[9] = {
        {  12,   12,   32,   16, conv1_w, conv1_b, bn1g, bn1b, bn1m, bn1v, 0, 0, 0},
        {  12,   64,   32,   64, conv2_w, conv2_b, bn2g, bn2b, bn2m, bn2v, 0, 0, 0},
        {  64,   64,   64,   64, conv3_w, conv3_b, bn3g, bn3b, bn3m, bn3v, 0, 0, 0},
        {  64,   64,   64,   64, g1lw,    g1lb,    0, 0, 0, 0, 0, 0, 0},
        {  64,  128,   64,  128, g1cw,    g1cb,    0, 0, 0, 0, 0, 0, 0},
        { 128,  128,  128,  128, g2lw,    g2lb,    0, 0, 0, 0, 0, 0, 0},
        { 128, 1024,  128, 1024, g2cw,    g2cb,    0, 0, 0, 0, 0, 0, 0},
        {1024, 1024, 1024, 1024, conv4_w, conv4_b, 0, 0, 0, 0, 0, 0, 0},
        {1024,  512, 1024,  512, conv5_w, conv5_b, 0, 0, 0, 0, 0, 0, 0},
    };
    for (int i = 0; i < 9; ++i) {
        ly[i].wt = (_Float16*)carve((size_t)ly[i].npad * ly[i].kpad * 2);
        ly[i].sc = (float*)carve((size_t)ly[i].npad * 4);
        ly[i].sh = (float*)carve((size_t)ly[i].npad * 4);
    }

    // ---- prep: weight transpose/pad + folded scale/shift ----
    for (int i = 0; i < 9; ++i) {
        int tot = ly[i].npad * ly[i].kpad;
        prep_wt<<<(tot + 255) / 256, 256, 0, stream>>>(ly[i].W, ly[i].wt,
            ly[i].K, ly[i].N, ly[i].kpad, ly[i].npad);
        prep_ss<<<(ly[i].npad + 255) / 256, 256, 0, stream>>>(ly[i].bias,
            ly[i].g, ly[i].b, ly[i].m, ly[i].v, ly[i].sc, ly[i].sh,
            ly[i].N, ly[i].npad);
    }

    const int knnBlocks = BATCH * (NPTS / 16);

    // ---- stage 0: knn on x (padded f16), covariance -> t0[BN][32] ----
    cvtpad_kernel<<<(BN * 32 + 255) / 256, 256, 0, stream>>>(x, hH, 3, 32);
    sq_kernel<<<(BN + 255) / 256, 256, 0, stream>>>(x, sqv, BN, 3);
    knn_kernel<32><<<knnBlocks, 32, 0, stream>>>(hH, sqv, idx);
    cov_kernel<<<(BN + 255) / 256, 256, 0, stream>>>(x, idx, t0);

    // ---- conv1/2/3 (BN+bias folded, relu) ----
    wmma_gemm<1><<<dim3(BN / 16, 1), 32, 0, stream>>>(t0, 32, ly[0].wt, 32,
        ly[0].sc, ly[0].sh, t1, 32, nullptr, 1, 1, 0, 16);
    wmma_gemm<4><<<dim3(BN / 16, 1), 32, 0, stream>>>(t1, 32, ly[1].wt, 32,
        ly[1].sc, ly[1].sh, t0, 64, nullptr, 1, 1, 0, 64);
    wmma_gemm<4><<<dim3(BN / 16, 1), 32, 0, stream>>>(t0, 64, ly[2].wt, 64,
        ly[2].sc, ly[2].sh, t1, 64, nullptr, 2, 1, 0, 64);

    // ---- graph layer 1: knn(C=64) -> gather-max -> 64 -> 128 relu ----
    cvtpad_kernel<<<(BN * 64 + 255) / 256, 256, 0, stream>>>(t1, hH, 64, 64);
    sq_kernel<<<(BN + 255) / 256, 256, 0, stream>>>(t1, sqv, BN, 64);
    knn_kernel<64><<<knnBlocks, 32, 0, stream>>>(hH, sqv, idx);
    gmax_kernel<<<(BN * 64 + 255) / 256, 256, 0, stream>>>(t1, idx, t0, 64);
    wmma_gemm<4><<<dim3(BN / 16, 1), 32, 0, stream>>>(t0, 64, ly[3].wt, 64,
        ly[3].sc, ly[3].sh, t2, 64, nullptr, 2, 0, 0, 64);
    wmma_gemm<4><<<dim3(BN / 16, 2), 32, 0, stream>>>(t2, 64, ly[4].wt, 64,
        ly[4].sc, ly[4].sh, t1, 128, nullptr, 2, 1, 0, 128);

    // ---- graph layer 2: knn(C=128) -> gather-max -> 128 -> 1024 + maxpool ----
    cvtpad_kernel<<<(BN * 128 + 255) / 256, 256, 0, stream>>>(t1, hH, 128, 128);
    sq_kernel<<<(BN + 255) / 256, 256, 0, stream>>>(t1, sqv, BN, 128);
    knn_kernel<128><<<knnBlocks, 32, 0, stream>>>(hH, sqv, idx);
    gmax_kernel<<<(BN * 128 + 255) / 256, 256, 0, stream>>>(t1, idx, t0, 128);
    wmma_gemm<4><<<dim3(BN / 16, 2), 32, 0, stream>>>(t0, 128, ly[5].wt, 128,
        ly[5].sc, ly[5].sh, t2, 128, nullptr, 4, 0, 0, 128);
    fill_kernel<<<(BATCH * 1024 + 255) / 256, 256, 0, stream>>>(pool, -3.4e38f, BATCH * 1024);
    wmma_gemm<4><<<dim3(BN / 16, 16), 32, 0, stream>>>(t2, 128, ly[6].wt, 128,
        ly[6].sc, ly[6].sh, nullptr, 1024, pool, 4, 0, NPTS, 1024);

    // ---- head: conv4 (relu) + conv5 -> d_out [16,1,512] ----
    wmma_gemm<4><<<dim3(1, 16), 32, 0, stream>>>(pool, 1024, ly[7].wt, 1024,
        ly[7].sc, ly[7].sh, t2, 1024, nullptr, 32, 1, 0, 1024);
    wmma_gemm<4><<<dim3(1, 8), 32, 0, stream>>>(t2, 1024, ly[8].wt, 1024,
        ly[8].sc, ly[8].sh, (float*)d_out, 512, nullptr, 32, 0, 0, 512);
}
